// Attention_83674552860687
// MI455X (gfx1250) — compile-verified, matrix operand order checked by reading, not run on hardware
//
#include <hip/hip_runtime.h>
#include <hip/hip_bf16.h>

// ---------------------------------------------------------------------------
// LLaMA-style attention block for MI455X (gfx1250, wave32, WMMA + TDM).
//   q = x@wq^T ; k = x@wk^T ; v = x@wv^T   (f16 WMMA, f32 accum)
//   RoPE(q,k) ; flash-attention w/ causal mask (f16 WMMA, f32 accum)
//   out = y@wo^T (f32 output)
// Tile staging uses the Tensor Data Mover (tensor_load_to_lds) with
// double-buffered LDS and hardware row padding; waits via s_wait_tensorcnt.
// ---------------------------------------------------------------------------

typedef __attribute__((ext_vector_type(16))) _Float16 v16h;
typedef __attribute__((ext_vector_type(8)))  float    v8f;
typedef unsigned int u32x4 __attribute__((ext_vector_type(4)));
typedef int          i32x8 __attribute__((ext_vector_type(8)));
typedef int          i32x4 __attribute__((ext_vector_type(4)));

#define S_LEN      2048
#define DIM        4096
#define KV_DIM     1024
#define HEAD_DIM   128
#define N_HEADS    32
#define N_KV_HEADS 8

// ---- TDM: 2-D tile (f16 elements) global -> LDS, with row padding ---------
// pad_interval_code: pad inserted after 2^(code+1) DWORDs; pad_amount fixed at
// 4 DWORDs (8 halves) -> LDS row stride = tile_w + 8 halves.
__device__ __forceinline__ void tdm_load_2d(uint32_t lds_byte_off,
                                            const void* gptr,
                                            uint32_t tile_w, uint32_t tile_h,
                                            uint32_t stride_elems,
                                            uint32_t pad_interval_code) {
  const uint64_t ga = (uint64_t)(uintptr_t)gptr;
  u32x4 g0;
  g0[0] = 1u;                                        // count=1, no gather
  g0[1] = lds_byte_off;                              // lds_addr
  g0[2] = (uint32_t)ga;                              // global_addr[31:0]
  g0[3] = (uint32_t)((ga >> 32) & 0x01ffffffu)       // global_addr[56:32]
          | 0x80000000u;                             // type=2 (image)
  i32x8 g1;
  g1[0] = (int)((1u << 16)                           // data_size = 2 bytes
                | (1u << 20)                         // pad_enable
                | (pad_interval_code << 22)          // pad interval
                | (3u << 25));                       // pad amount = 4 dwords
  g1[1] = (int)((tile_w & 0xffffu) << 16);           // tensor_dim0[15:0]
  g1[2] = (int)(((tile_w >> 16) & 0xffffu)           // tensor_dim0[31:16]
                | ((tile_h & 0xffffu) << 16));       // tensor_dim1[15:0]
  g1[3] = (int)(((tile_h >> 16) & 0xffffu)           // tensor_dim1[31:16]
                | ((tile_w & 0xffffu) << 16));       // tile_dim0
  g1[4] = (int)(tile_h & 0xffffu);                   // tile_dim1 (tile_dim2=0)
  g1[5] = (int)stride_elems;                         // tensor_dim0_stride[31:0]
  g1[6] = 0;                                         // stride[47:32], dim1_stride
  g1[7] = 0;
  const i32x4 z4 = {0, 0, 0, 0};
  const i32x8 z8 = {0, 0, 0, 0, 0, 0, 0, 0};
  // 6-arg form (clang-23 / therock headers): groups 0..3 + extra group + cpol
  __builtin_amdgcn_tensor_load_to_lds(g0, g1, z4, z4, z8, 0);
}

// ---- WMMA fragment loaders (CDNA5 ISA 7.12.2, 16-bit 16x32 A layout) ------
__device__ __forceinline__ v16h frag_load_row(const _Float16* __restrict__ base,
                                              int ld) {
  const int lane = threadIdx.x & 31;
  const int r    = lane & 15;
  const int hf   = lane >> 4;
  const _Float16* p = base + r * ld + hf * 8;
  v16h f;
#pragma unroll
  for (int jj = 0; jj < 8; ++jj) {
    const int k = (jj < 4) ? (2 * jj) : (2 * jj + 8);  // 0,2,4,6,16,18,20,22
    f[2 * jj]     = p[k];
    f[2 * jj + 1] = p[k + 1];
  }
  return f;
}

// B operand from a K-major tile: element B[k][n] at base[k*ld + n].
__device__ __forceinline__ v16h frag_load_col(const _Float16* __restrict__ base,
                                              int ld) {
  const int lane = threadIdx.x & 31;
  const int r    = lane & 15;
  const int hf   = lane >> 4;
  v16h f;
#pragma unroll
  for (int j = 0; j < 16; ++j) {
    const int k = ((j & 8) ? (j + 8) : j) + hf * 8;
    f[j] = base[k * ld + r];
  }
  return f;
}

// ---- f32 -> f16 cast ------------------------------------------------------
__global__ void cast_f32_to_f16(const float* __restrict__ in,
                                _Float16* __restrict__ out, int n) {
  const int i = blockIdx.x * blockDim.x + threadIdx.x;
  if (i < n) out[i] = (_Float16)in[i];
}

// ---- tiled GEMM: C[M,N] = A[M,K] * B[N,K]^T  (TDM-staged, double-buffered) -
template <typename OutT>
__global__ __launch_bounds__(256)
void gemm_nt(const _Float16* __restrict__ A,
             const _Float16* __restrict__ B,
             OutT* __restrict__ C,
             int M, int N, int Kd) {
  constexpr int BM = 128, BN = 128, BK = 64, LDT = BK + 8;   // 144B rows
  extern __shared__ char smem_raw[];
  _Float16* As = (_Float16*)smem_raw;          // [2][BM][LDT]
  _Float16* Bs = As + 2 * BM * LDT;            // [2][BN][LDT]
  constexpr uint32_t tileBytes = BM * LDT * 2; // 18,432 B
  const uint32_t ofsA = 0;
  const uint32_t ofsB = 2u * tileBytes;

  const int bm  = blockIdx.y * BM;
  const int bn  = blockIdx.x * BN;
  const int tid = threadIdx.x;
  const int w   = tid >> 5;   // 8 waves
  const int wr  = w >> 2;     // 0..1 -> 64 rows
  const int wc  = w & 3;      // 0..3 -> 32 cols
  const int NT  = Kd / BK;

  v8f acc[4][2] = {};

  if (w == 0) {  // prologue: DMA tile 0 into buffer 0
    tdm_load_2d(ofsA, A + (size_t)bm * Kd, BK, BM, Kd, 4);
    tdm_load_2d(ofsB, B + (size_t)bn * Kd, BK, BN, Kd, 4);
  }

  for (int kt = 0; kt < NT; ++kt) {
    const int buf = kt & 1;
    if (w == 0) {
      if (kt + 1 < NT) {  // prefetch next tile, then wait only for current
        const size_t koff = (size_t)(kt + 1) * BK;
        tdm_load_2d(ofsA + (uint32_t)(buf ^ 1) * tileBytes,
                    A + (size_t)bm * Kd + koff, BK, BM, Kd, 4);
        tdm_load_2d(ofsB + (uint32_t)(buf ^ 1) * tileBytes,
                    B + (size_t)bn * Kd + koff, BK, BN, Kd, 4);
        __builtin_amdgcn_s_wait_tensorcnt(2);
      } else {
        __builtin_amdgcn_s_wait_tensorcnt(0);
      }
    }
    __syncthreads();

    const _Float16* Ab = As + buf * (BM * LDT);
    const _Float16* Bb = Bs + buf * (BN * LDT);
#pragma unroll
    for (int kk = 0; kk < 2; ++kk) {      // two 32-deep WMMA k-steps
      v16h bf[2];
#pragma unroll
      for (int j = 0; j < 2; ++j)
        bf[j] = frag_load_row(Bb + (wc * 32 + j * 16) * LDT + kk * 32, LDT);
#pragma unroll
      for (int i = 0; i < 4; ++i) {
        v16h af = frag_load_row(Ab + (wr * 64 + i * 16) * LDT + kk * 32, LDT);
#pragma unroll
        for (int j = 0; j < 2; ++j)
          acc[i][j] = __builtin_amdgcn_wmma_f32_16x16x32_f16(
              false, af, false, bf[j], (short)0, acc[i][j], false, false);
      }
    }
    __syncthreads();   // compute done before buffer is DMA-overwritten
  }

  const int lane = tid & 31;
  const int r    = lane & 15;
  const int hf   = lane >> 4;
#pragma unroll
  for (int i = 0; i < 4; ++i)
#pragma unroll
    for (int j = 0; j < 2; ++j) {
      const int n = bn + wc * 32 + j * 16 + r;
#pragma unroll
      for (int e = 0; e < 8; ++e) {
        const int m = bm + wr * 64 + i * 16 + e + hf * 8;
        C[(size_t)m * N + n] = (OutT)acc[i][j][e];
      }
    }
}

// ---- RoPE (interleaved pairs, freqs_cis = [S][64][2] f32) -----------------
__global__ void rope_apply(_Float16* __restrict__ qk,
                           const float* __restrict__ fc, int nheads) {
  const int idx   = blockIdx.x * blockDim.x + threadIdx.x;
  const int total = S_LEN * nheads * 64;
  if (idx >= total) return;
  const int p  = idx & 63;
  const int hh = (idx >> 6) % nheads;
  const int s  = idx / (64 * nheads);
  const float c  = fc[(s * 64 + p) * 2 + 0];
  const float sn = fc[(s * 64 + p) * 2 + 1];
  _Float16* b = qk + ((size_t)s * nheads + hh) * HEAD_DIM + p * 2;
  const float x0 = (float)b[0];
  const float x1 = (float)b[1];
  b[0] = (_Float16)(x0 * c - x1 * sn);
  b[1] = (_Float16)(x1 * c + x0 * sn);
}

// ---- flash attention: 4 waves x 16 q-rows, TDM double-buffered K/V --------
__global__ __launch_bounds__(128)
void attn_fa(const _Float16* __restrict__ Q,   // [S, 32*128]
             const _Float16* __restrict__ K,   // [S,  8*128]
             const _Float16* __restrict__ V,   // [S,  8*128]
             _Float16* __restrict__ Y) {       // [S, 32*128]
  constexpr int LDK = HEAD_DIM + 8;            // 272B rows
  extern __shared__ char smem_raw[];
  _Float16* Ks = (_Float16*)smem_raw;          // [2][64][LDK]
  _Float16* Vs = Ks + 2 * 64 * LDK;            // [2][64][LDK]
  _Float16* Ps = Vs + 2 * 64 * LDK;            // [4][16][72]
  constexpr uint32_t kvTileBytes = 64 * LDK * 2;   // 17,408 B
  const uint32_t ofsK = 0;
  const uint32_t ofsV = 2u * kvTileBytes;

  const int h    = blockIdx.y;
  const int kvh  = h >> 2;                     // 32 heads / 8 kv heads
  const int qt   = blockIdx.x;                 // 64-row q tile
  const int tid  = threadIdx.x;
  const int wid  = tid >> 5;
  const int lane = tid & 31;
  const int r    = lane & 15;
  const int hf   = lane >> 4;
  const int q0   = qt * 64 + wid * 16;
  _Float16* Pw   = Ps + wid * 16 * 72;

  v16h qfrag[4];
#pragma unroll
  for (int f = 0; f < 4; ++f)
    qfrag[f] = frag_load_row(Q + (size_t)q0 * DIM + h * HEAD_DIM + f * 32, DIM);

  v8f  o[8] = {};
  float mrow[8], lrow[8];
#pragma unroll
  for (int i = 0; i < 8; ++i) { mrow[i] = -1e30f; lrow[i] = 0.f; }

  const float scale = 0.08838834764831845f;    // 1/sqrt(128)

  if (wid == 0) {  // prologue DMA: K/V tile 0 -> buffer 0
    tdm_load_2d(ofsK, K + (size_t)kvh * HEAD_DIM, HEAD_DIM, 64, KV_DIM, 5);
    tdm_load_2d(ofsV, V + (size_t)kvh * HEAD_DIM, HEAD_DIM, 64, KV_DIM, 5);
  }

  for (int kt = 0; kt <= qt; ++kt) {           // causal: only needed k tiles
    const int buf = kt & 1;
    if (wid == 0) {
      if (kt + 1 <= qt) {
        const size_t ro = (size_t)(kt + 1) * 64 * KV_DIM + kvh * HEAD_DIM;
        tdm_load_2d(ofsK + (uint32_t)(buf ^ 1) * kvTileBytes, K + ro,
                    HEAD_DIM, 64, KV_DIM, 5);
        tdm_load_2d(ofsV + (uint32_t)(buf ^ 1) * kvTileBytes, V + ro,
                    HEAD_DIM, 64, KV_DIM, 5);
        __builtin_amdgcn_s_wait_tensorcnt(2);
      } else {
        __builtin_amdgcn_s_wait_tensorcnt(0);
      }
    }
    __syncthreads();
    const _Float16* Kb = Ks + buf * (64 * LDK);
    const _Float16* Vb = Vs + buf * (64 * LDK);

    // scores: S[16x64] = Q[16x128] * K_tile[64x128]^T
    v8f sc[4] = {};
#pragma unroll
    for (int nt = 0; nt < 4; ++nt)
#pragma unroll
      for (int f = 0; f < 4; ++f) {
        v16h kf = frag_load_row(Kb + (nt * 16) * LDK + f * 32, LDK);
        sc[nt] = __builtin_amdgcn_wmma_f32_16x16x32_f16(
            false, qfrag[f], false, kf, (short)0, sc[nt], false, false);
      }

    // scale + causal mask + per-row running max
    float tmax[8];
#pragma unroll
    for (int i = 0; i < 8; ++i) tmax[i] = -1e30f;
#pragma unroll
    for (int nt = 0; nt < 4; ++nt) {
      const int n_g = kt * 64 + nt * 16 + r;
#pragma unroll
      for (int i = 0; i < 8; ++i) {
        const int m_g = q0 + i + hf * 8;
        float v = sc[nt][i] * scale + (n_g <= m_g ? 0.f : -1.0e9f);
        sc[nt][i] = v;
        tmax[i] = fmaxf(tmax[i], v);
      }
    }
#pragma unroll
    for (int i = 0; i < 8; ++i)
#pragma unroll
      for (int off = 8; off >= 1; off >>= 1)
        tmax[i] = fmaxf(tmax[i], __shfl_xor(tmax[i], off, 32));

    float alpha[8], rsum[8];
#pragma unroll
    for (int i = 0; i < 8; ++i) {
      const float mnew = fmaxf(mrow[i], tmax[i]);
      alpha[i] = __expf(mrow[i] - mnew);
      mrow[i]  = mnew;
      rsum[i]  = 0.f;
    }
    // p = exp(s - m); spill to LDS to re-layout as A fragments
#pragma unroll
    for (int nt = 0; nt < 4; ++nt)
#pragma unroll
      for (int i = 0; i < 8; ++i) {
        const float p = __expf(sc[nt][i] - mrow[i]);
        rsum[i] += p;
        Pw[(i + hf * 8) * 72 + nt * 16 + r] = (_Float16)p;
      }
#pragma unroll
    for (int i = 0; i < 8; ++i) {
#pragma unroll
      for (int off = 8; off >= 1; off >>= 1)
        rsum[i] += __shfl_xor(rsum[i], off, 32);
      lrow[i] = lrow[i] * alpha[i] + rsum[i];
#pragma unroll
      for (int j = 0; j < 8; ++j) o[j][i] *= alpha[i];
    }
    __syncthreads();   // P visible (all waves run identical trip counts)

    // O[16x128] += P[16x64] * V[64x128]
#pragma unroll
    for (int kk = 0; kk < 2; ++kk) {
      v16h pf = frag_load_row(Pw + kk * 32, 72);
#pragma unroll
      for (int j = 0; j < 8; ++j) {
        v16h vf = frag_load_col(Vb + (kk * 32) * LDK + j * 16, LDK);
        o[j] = __builtin_amdgcn_wmma_f32_16x16x32_f16(
            false, pf, false, vf, (short)0, o[j], false, false);
      }
    }
    __syncthreads();   // protect K/V buffer before DMA reuse
  }

  // normalize and store
#pragma unroll
  for (int i = 0; i < 8; ++i) {
    const float inv = 1.f / lrow[i];
    const int m_g = q0 + i + hf * 8;
#pragma unroll
    for (int j = 0; j < 8; ++j) {
      const int col = h * HEAD_DIM + j * 16 + r;
      Y[(size_t)m_g * DIM + col] = (_Float16)(o[j][i] * inv);
    }
  }
}

// ---------------------------------------------------------------------------
extern "C" void kernel_launch(void* const* d_in, const int* in_sizes, int n_in,
                              void* d_out, int out_size, void* d_ws,
                              size_t ws_size, hipStream_t stream) {
  (void)in_sizes; (void)n_in; (void)out_size; (void)ws_size;

  const float* x  = (const float*)d_in[0];
  const float* fc = (const float*)d_in[1];   // freqs_cis
  // d_in[2] = mask (unused; causal mask applied analytically)
  const float* wq = (const float*)d_in[3];
  const float* wk = (const float*)d_in[4];
  const float* wv = (const float*)d_in[5];
  const float* wo = (const float*)d_in[6];
  float* out = (float*)d_out;

  char* ws = (char*)d_ws;
  size_t off = 0;
  auto bump = [&](size_t elems) {
    _Float16* p = (_Float16*)(ws + off);
    off += elems * sizeof(_Float16);
    return p;
  };
  _Float16* xh  = bump((size_t)S_LEN * DIM);
  _Float16* wqh = bump((size_t)DIM * DIM);
  _Float16* wkh = bump((size_t)KV_DIM * DIM);
  _Float16* wvh = bump((size_t)KV_DIM * DIM);
  _Float16* woh = bump((size_t)DIM * DIM);
  _Float16* Qh  = bump((size_t)S_LEN * DIM);
  _Float16* Kh  = bump((size_t)S_LEN * KV_DIM);
  _Float16* Vh  = bump((size_t)S_LEN * KV_DIM);
  _Float16* Yh  = bump((size_t)S_LEN * DIM);

  auto cast = [&](const float* src, _Float16* dst, size_t n) {
    cast_f32_to_f16<<<(int)((n + 255) / 256), 256, 0, stream>>>(src, dst,
                                                                (int)n);
  };
  cast(x,  xh,  (size_t)S_LEN * DIM);
  cast(wq, wqh, (size_t)DIM * DIM);
  cast(wk, wkh, (size_t)KV_DIM * DIM);
  cast(wv, wvh, (size_t)KV_DIM * DIM);
  cast(wo, woh, (size_t)DIM * DIM);

  const int gemmLds = 4 * 128 * 72 * 2;           // 73,728 B
  const int attnLds = (2 * 64 * 136 * 2) * 2      // K+V double buffers
                      + 4 * 16 * 72 * 2;          // + P tiles = 78,848 B
  (void)hipFuncSetAttribute((const void*)gemm_nt<_Float16>,
                            hipFuncAttributeMaxDynamicSharedMemorySize,
                            gemmLds);
  (void)hipFuncSetAttribute((const void*)gemm_nt<float>,
                            hipFuncAttributeMaxDynamicSharedMemorySize,
                            gemmLds);
  (void)hipFuncSetAttribute((const void*)attn_fa,
                            hipFuncAttributeMaxDynamicSharedMemorySize,
                            attnLds);

  gemm_nt<_Float16><<<dim3(DIM / 128, S_LEN / 128), 256, gemmLds, stream>>>(
      xh, wqh, Qh, S_LEN, DIM, DIM);
  gemm_nt<_Float16><<<dim3(KV_DIM / 128, S_LEN / 128), 256, gemmLds, stream>>>(
      xh, wkh, Kh, S_LEN, KV_DIM, DIM);
  gemm_nt<_Float16><<<dim3(KV_DIM / 128, S_LEN / 128), 256, gemmLds, stream>>>(
      xh, wvh, Vh, S_LEN, KV_DIM, DIM);

  rope_apply<<<(S_LEN * N_HEADS * 64 + 255) / 256, 256, 0, stream>>>(
      Qh, fc, N_HEADS);
  rope_apply<<<(S_LEN * N_KV_HEADS * 64 + 255) / 256, 256, 0, stream>>>(
      Kh, fc, N_KV_HEADS);

  attn_fa<<<dim3(S_LEN / 64, N_HEADS), 128, attnLds, stream>>>(Qh, Kh, Vh, Yh);

  gemm_nt<float><<<dim3(DIM / 128, S_LEN / 128), 256, gemmLds, stream>>>(
      Yh, woh, out, S_LEN, DIM, DIM);
}
